// BioBERT_ARG_GNN_2233382994519
// MI455X (gfx1250) — matrix operand order repaired and checked by reading, not compiled, not run on hardware
//
#include <hip/hip_runtime.h>
#include <hip/hip_bf16.h>
#include <math.h>

// ---------------- compile-time problem shape ----------------
#define Bn   64
#define Sn   512
#define Hn   768
#define Nn   128          // nodes per graph
#define En   1024         // edges per graph
#define GHn  128          // GCN hidden
#define FHn  256          // head hidden
#define Ln   2
#define BN   (Bn * Nn)    // 8192 total nodes
#define ET   (Bn * En)    // 65536 total edges
#define NT   8            // N-tiles in 128-wide output (8 waves/block)

typedef __attribute__((ext_vector_type(16))) _Float16 v16h;
typedef __attribute__((ext_vector_type(2)))  _Float16 v2h;
typedef __attribute__((ext_vector_type(8)))  float    v8f;

// ---------------- helpers ----------------
__device__ __forceinline__ float blockReduceSum256(float v, float* sdata) {
    // wave32 reduce, then cross-wave via LDS (blockDim.x == 256 -> 8 waves)
    int lane = threadIdx.x & 31;
    int wid  = threadIdx.x >> 5;
#pragma unroll
    for (int off = 16; off > 0; off >>= 1) v += __shfl_xor(v, off, 32);
    if (lane == 0) sdata[wid] = v;
    __syncthreads();
    if (wid == 0) {
        float x = (lane < 8) ? sdata[lane] : 0.0f;
#pragma unroll
        for (int off = 4; off > 0; off >>= 1) x += __shfl_xor(x, off, 32);
        if (lane == 0) sdata[0] = x;
    }
    __syncthreads();
    float r = sdata[0];
    __syncthreads();
    return r;
}

// ---------------- 1) fused sigmoid-gate + segment mean-pool ----------------
// One block per (batch, node). Streams last_hidden exactly once overall.
__global__ void k_gate_pool(const float* __restrict__ lh,
                            const int*   __restrict__ submap,
                            const float* __restrict__ wr,
                            const float* __restrict__ brp,
                            float*       __restrict__ nf) {
    const int n = blockIdx.x;      // node id
    const int b = blockIdx.y;      // batch id
    const int t = threadIdx.x;     // 0..255

    __shared__ int   smap[Sn];
    __shared__ float sdata[32];

    for (int i = t; i < Sn; i += 256) smap[i] = submap[b * Sn + i];
    __syncthreads();

    const float wr0 = wr[t], wr1 = wr[t + 256], wr2 = wr[t + 512];
    const float br  = brp[0];

    float a0 = 0.f, a1 = 0.f, a2 = 0.f;
    int   cnt = 0;
    const float* lhb = lh + (size_t)b * Sn * Hn;

    for (int s = 0; s < Sn; ++s) {
        if (smap[s] != n) continue;                 // uniform across block
        const float* row = lhb + (size_t)s * Hn;
        __builtin_prefetch(row + Hn, 0, 0);         // global_prefetch_b8 (speculative)
        const float l0 = row[t], l1 = row[t + 256], l2 = row[t + 512];
        const float part = l0 * wr0 + l1 * wr1 + l2 * wr2;
        const float z = blockReduceSum256(part, sdata) + br;
        const float g = 1.0f / (1.0f + __expf(-z));
        a0 += l0 * g; a1 += l1 * g; a2 += l2 * g;
        ++cnt;
    }

    const float inv = 1.0f / fmaxf((float)cnt, 1.0f);
    float* o = nf + ((size_t)b * Nn + n) * Hn;
    o[t]       = a0 * inv;
    o[t + 256] = a1 * inv;
    o[t + 512] = a2 * inv;
}

// ---------------- 2) degree / normalization ----------------
__global__ void k_deg_init(float* __restrict__ deg) {
    int i = blockIdx.x * blockDim.x + threadIdx.x;
    if (i < BN) deg[i] = 1.0f;                      // self-loop contribution
}

__global__ void k_deg_scatter(const int* __restrict__ edge_index,
                              float* __restrict__ deg) {
    int tid = blockIdx.x * blockDim.x + threadIdx.x;
    if (tid >= ET) return;
    int b = tid / En, e = tid % En;
    const int* ei = edge_index + (size_t)b * 2 * En;
    int d = ei[En + e] + b * Nn;                    // edge_index[b,1,e]
    atomicAdd(&deg[d], 1.0f);
}

__global__ void k_dinv(float* __restrict__ deg) {
    int i = blockIdx.x * blockDim.x + threadIdx.x;
    if (i < BN) deg[i] = rsqrtf(deg[i]);            // in-place deg -> dinv
}

// ---------------- 3a) pre-pack W (K x 128 fp32) into f16 WMMA-B layout ------
// Bp[(((kt*NT + nt)*32 + lane)*16 + j] = W[(kt*32 + kk)*128 + nt*16 + (lane&15)]
// where kk follows the ISA 16-bit 32x16 B-fragment K mapping per lane half.
__global__ void k_pack_b(const float* __restrict__ W,
                         _Float16* __restrict__ Bp, int K) {
    int tid = blockIdx.x * blockDim.x + threadIdx.x;
    if (tid >= K * 128) return;                     // (K/32)*8*32*16 == K*128
    int j    = tid & 15;
    int lane = (tid >> 4) & 31;
    int nt   = (tid >> 9) & 7;
    int kt   = tid >> 12;
    int kk   = (j < 8) ? ((lane >> 4) * 8 + j) : (16 + (lane >> 4) * 8 + (j - 8)); // B frag: K=8h+j (j<8), 16+8h+(j-8) (j>=8)
    int col  = nt * 16 + (lane & 15);
    Bp[tid] = (_Float16)W[(size_t)(kt * 32 + kk) * GHn + col];
}

// ---------------- 3b) WMMA GEMM: C[M,128] = A[M,K] @ W[K,128] ----------------
// Block = 8 waves sharing one 16-row M-tile. A tile staged in LDS as f16 in
// the exact A-fragment lane layout (double buffered). Staging is straight-line:
// each of the 256 threads moves one aligned f16 pair (global_load_b64 ->
// v_cvt_pk_f16_f32 -> ds_store_b32). Per K-step each wave then does
// 2x ds_load_b128 (A) + 2x global_load_b128 (packed B) + 1 WMMA.
__global__ void k_gemm_wmma(const float* __restrict__ A,
                            const _Float16* __restrict__ Bp,
                            float* __restrict__ C, int K) {
    __shared__ alignas(32) _Float16 As16[2][32 * 16];   // 2 x 1KB, fragment layout

    const int m0   = blockIdx.x * 16;               // gridDim.x = M/16
    const int lane = threadIdx.x & 31;
    const int nt   = threadIdx.x >> 5;              // wave id == N-tile (0..7)
    const int KT   = K >> 5;                        // K-steps of 32

    // Per-thread staging slot (constant across K-steps):
    //   thread p stages pair j = 2*(p&7) of lane-slot l = p>>3.
    //   A-frag K map: l<16: K = (j<8? j : j+8);  l>=16: K = (j<8? j+8 : j+16)
    //   (j even => both halves of the pair lie in the same contiguous K-run)
    const int sl  = threadIdx.x >> 3;               // lane slot 0..31
    const int sj  = (threadIdx.x & 7) << 1;         // pair start 0,2,..,14
    const int skk = (sl < 16) ? ((sj < 8) ? sj : sj + 8)
                              : ((sj < 8) ? sj + 8 : sj + 16);
    const float* srow = A + (size_t)(m0 + (sl & 15)) * K + skk;

    auto stage = [&](int kt, int buf) {
        float2 v = *(const float2*)(srow + (kt << 5));
        v2h h;
        h[0] = (_Float16)v.x;
        h[1] = (_Float16)v.y;
        *(v2h*)(&As16[buf][sl * 16 + sj]) = h;
    };

    v8f acc = {};

    stage(0, 0);
    __syncthreads();

    for (int kt = 0; kt < KT; ++kt) {
        const int cur = kt & 1;

        // A fragment: one 32B aligned LDS slot per lane (2x ds_load_b128)
        v16h a = *(const v16h*)(&As16[cur][lane * 16]);
        // B fragment: contiguous 32B per lane from the pre-packed weights
        v16h bf = *(const v16h*)(Bp + (((size_t)kt * NT + nt) * 32 + lane) * 16);

        if (kt + 1 < KT) stage(kt + 1, cur ^ 1);    // overlap copy with WMMA

        acc = __builtin_amdgcn_wmma_f32_16x16x32_f16(
            /*neg_a=*/false, a, /*neg_b=*/false, bf,
            /*c_mod=*/(short)0, acc, /*reuse_a=*/false, /*reuse_b=*/false);

        __syncthreads();
    }

    // store per C/D layout: VGPR r -> row m0 + r + (lane>=16 ? 8 : 0)
    const int col = nt * 16 + (lane & 15);
    const int mo  = m0 + ((lane >= 16) ? 8 : 0);
#pragma unroll
    for (int r = 0; r < 8; ++r)
        C[(size_t)(mo + r) * GHn + col] = acc[r];
}

// ---------------- 4) GCN edge aggregation ----------------
__global__ void k_agg_init(const float* __restrict__ h,
                           const float* __restrict__ dinv,
                           float* __restrict__ a) {
    int tid = blockIdx.x * blockDim.x + threadIdx.x;
    if (tid >= BN * GHn) return;
    int node = tid / GHn;
    float w = dinv[node];
    a[tid] = h[tid] * w * w;                        // self-loop term
}

__global__ void k_agg_scatter(const float* __restrict__ h,
                              const int*   __restrict__ edge_index,
                              const float* __restrict__ dinv,
                              float* __restrict__ a) {
    int tid  = blockIdx.x * blockDim.x + threadIdx.x;
    int edge = tid >> 5;
    int lane = tid & 31;
    if (edge >= ET) return;
    int b = edge / En, e = edge % En;
    const int* ei = edge_index + (size_t)b * 2 * En;
    int s = ei[e]      + b * Nn;
    int d = ei[En + e] + b * Nn;
    float w = dinv[s] * dinv[d];
    const float* hs = h + (size_t)s * GHn;
    float*       ad = a + (size_t)d * GHn;
#pragma unroll
    for (int c = lane; c < GHn; c += 32)
        atomicAdd(&ad[c], hs[c] * w);
}

__global__ void k_bias_relu(const float* __restrict__ a,
                            const float* __restrict__ bias,
                            float* __restrict__ x) {
    int tid = blockIdx.x * blockDim.x + threadIdx.x;
    if (tid >= BN * GHn) return;
    int c = tid & (GHn - 1);
    x[tid] = fmaxf(a[tid] + bias[c], 0.0f);
}

// ---------------- 5) mean pool over nodes ----------------
__global__ void k_pool(const float* __restrict__ x, float* __restrict__ pooled) {
    int b = blockIdx.x;          // 0..63
    int c = threadIdx.x;         // 0..127
    float s = 0.f;
    const float* xb = x + (size_t)b * Nn * GHn;
    for (int n = 0; n < Nn; ++n) s += xb[(size_t)n * GHn + c];
    pooled[b * GHn + c] = s * (1.0f / (float)Nn);
}

// ---------------- 6) classification head ----------------
__global__ void k_head(const float* __restrict__ lh,
                       const float* __restrict__ pooled,
                       const float* __restrict__ Wf1,
                       const float* __restrict__ bf1,
                       const float* __restrict__ Wf2,
                       const float* __restrict__ bf2,
                       float* __restrict__ out) {
    const int b = blockIdx.x;
    const int t = threadIdx.x;   // 0..255, one hidden unit each
    __shared__ float cat[Hn + GHn];
    __shared__ float sdata[32];

    for (int i = t; i < Hn; i += 256) cat[i] = lh[((size_t)b * Sn) * Hn + i]; // cls token
    for (int i = t; i < GHn; i += 256) cat[Hn + i] = pooled[b * GHn + i];
    __syncthreads();

    float acc = bf1[t];
    for (int k = 0; k < Hn + GHn; ++k) acc += cat[k] * Wf1[(size_t)k * FHn + t];
    float h = fmaxf(acc, 0.0f);

    float r0 = blockReduceSum256(h * Wf2[t * Ln + 0], sdata);
    float r1 = blockReduceSum256(h * Wf2[t * Ln + 1], sdata);
    if (t == 0) {
        out[b * Ln + 0] = r0 + bf2[0];
        out[b * Ln + 1] = r1 + bf2[1];
    }
}

// ---------------- launcher ----------------
extern "C" void kernel_launch(void* const* d_in, const int* in_sizes, int n_in,
                              void* d_out, int out_size, void* d_ws, size_t ws_size,
                              hipStream_t stream) {
    (void)in_sizes; (void)n_in; (void)out_size; (void)ws_size;

    const float* lh     = (const float*)d_in[0];
    const int*   submap = (const int*)  d_in[1];
    const int*   eidx   = (const int*)  d_in[2];
    /* d_in[3] = num_nodes (compile-time Nn) */
    const float* wr     = (const float*)d_in[4];
    const float* br     = (const float*)d_in[5];
    const float* W1     = (const float*)d_in[6];
    const float* b1     = (const float*)d_in[7];
    const float* W2     = (const float*)d_in[8];
    const float* b2     = (const float*)d_in[9];
    const float* Wf1    = (const float*)d_in[10];
    const float* bf1    = (const float*)d_in[11];
    const float* Wf2    = (const float*)d_in[12];
    const float* bf2    = (const float*)d_in[13];
    float* out = (float*)d_out;

    // workspace layout (floats); a/x alias the nf region (dead after GEMM1)
    float* ws     = (float*)d_ws;
    float* nf     = ws;                              // 8192*768
    float* hbuf   = ws + (size_t)BN * Hn;            // 8192*128
    float* dinv   = hbuf + (size_t)BN * GHn;         // 8192
    float* pooled = dinv + BN;                       // 8192
    _Float16* w1p = (_Float16*)(pooled + BN);        // 768*128 halfs (f16 packed W1)
    _Float16* w2p = w1p + (size_t)Hn * GHn;          // 128*128 halfs (f16 packed W2)
    float* abuf   = nf;                              // alias (8192*128)
    float* xbuf   = nf + (size_t)BN * GHn;           // alias (8192*128)

    // 0) one-time weight pre-pack into f16 WMMA-B layout
    k_pack_b<<<(Hn * GHn + 255) / 256, 256, 0, stream>>>(W1, w1p, Hn);
    k_pack_b<<<(GHn * GHn + 255) / 256, 256, 0, stream>>>(W2, w2p, GHn);

    // 1) fused gate + segment mean-pool (single pass over last_hidden)
    k_gate_pool<<<dim3(Nn, Bn), 256, 0, stream>>>(lh, submap, wr, br, nf);

    // 2) symmetric normalization
    k_deg_init   <<<(BN + 255) / 256, 256, 0, stream>>>(dinv);
    k_deg_scatter<<<(ET + 255) / 256, 256, 0, stream>>>(eidx, dinv);
    k_dinv       <<<(BN + 255) / 256, 256, 0, stream>>>(dinv);

    const int gemm_blocks = BN / 16;                 // 512 M-tiles
    const int elems = BN * GHn;

    // 3) GCN layer 1
    k_gemm_wmma  <<<gemm_blocks, 256, 0, stream>>>(nf, w1p, hbuf, Hn);
    k_agg_init   <<<(elems + 255) / 256, 256, 0, stream>>>(hbuf, dinv, abuf);
    k_agg_scatter<<<(ET * 32 + 255) / 256, 256, 0, stream>>>(hbuf, eidx, dinv, abuf);
    k_bias_relu  <<<(elems + 255) / 256, 256, 0, stream>>>(abuf, b1, xbuf);

    // 4) GCN layer 2
    k_gemm_wmma  <<<gemm_blocks, 256, 0, stream>>>(xbuf, w2p, hbuf, GHn);
    k_agg_init   <<<(elems + 255) / 256, 256, 0, stream>>>(hbuf, dinv, abuf);
    k_agg_scatter<<<(ET * 32 + 255) / 256, 256, 0, stream>>>(hbuf, eidx, dinv, abuf);
    k_bias_relu  <<<(elems + 255) / 256, 256, 0, stream>>>(abuf, b2, xbuf);

    // 5) pool + head
    k_pool<<<Bn, GHn, 0, stream>>>(xbuf, pooled);
    k_head<<<Bn, 256, 0, stream>>>(lh, pooled, Wf1, bf1, Wf2, bf2, out);
}
